// PointNet_10969346474599
// MI455X (gfx1250) — matrix operand order repaired and checked by reading, not compiled, not run on hardware
//
#include <hip/hip_runtime.h>
#include <hip/hip_bf16.h>

// ---------------------------------------------------------------------------
// PointNet head on MI455X (gfx1250, wave32, WMMA).
// GEMM chain in f16 via v_wmma_f32_16x16x32_f16 (A-operand reuse hints via
// template-unrolled tile loop), fused bias+ReLU epilogue, fused
// concat(context,f1) in the A-fragment loader (compile-time SPLIT),
// LDS-tiled kNN interpolation.
// ---------------------------------------------------------------------------

typedef _Float16 half_t;
typedef __attribute__((ext_vector_type(16))) _Float16 v16h;
typedef __attribute__((ext_vector_type(8)))  _Float16 v8h;
typedef __attribute__((ext_vector_type(8)))  float    v8f;

#define ROWS   32768   // B*N
#define NB     8       // batches
#define NPTS   4096    // points per batch

// ---------------------------------------------------------------------------
// Fragment loaders.
//
// 16-bit A matrix 16x32 (ISA 7.12.2): lanes 0-15 hold row M=lane with halves
// {0..7}->K k0+0..7, {8..15}->K k0+16..23; lanes 16-31 hold row M=lane-16 with
// halves {0..7}->K k0+8..15, {8..15}->K k0+24..31.  Two contiguous 16B loads.
//
// B matrix 32x16: lane holds column N=lane&15; same K order as A: lanes 0-15
// {0..7}->K k0+0..7, {8..15}->K k0+16..23; lanes 16-31 {0..7}->K k0+8..15,
// {8..15}->K k0+24..31.  Weights pre-transposed Wt[n][k]: two 16B loads.
// ---------------------------------------------------------------------------

template <bool SPLIT>
__device__ __forceinline__ v8h ldA8(const half_t* __restrict__ A2, int lda2,
                                    const half_t* __restrict__ A1, int lda1,
                                    int rshift, int ksplit, int row, int k) {
    const half_t* p;
    if (SPLIT && k < ksplit) p = A1 + (size_t)(row >> rshift) * lda1 + k;  // context
    else                     p = A2 + (size_t)row * lda2 + (SPLIT ? k - ksplit : k);
    return *(const v8h*)p;
}

template <bool SPLIT>
__device__ __forceinline__ v16h load_frag_a(const half_t* __restrict__ A2, int lda2,
                                            const half_t* __restrict__ A1, int lda1,
                                            int rshift, int ksplit,
                                            int row0, int k0) {
    int lane = threadIdx.x & 31;
    int hi   = lane >> 4;
    int row  = row0 + (lane & 15);
    int kb   = k0 + hi * 8;
    v8h lo = ldA8<SPLIT>(A2, lda2, A1, lda1, rshift, ksplit, row, kb);
    v8h hv = ldA8<SPLIT>(A2, lda2, A1, lda1, rshift, ksplit, row, kb + 16);
    v16h a;
#pragma unroll
    for (int i = 0; i < 8; ++i) { a[i] = lo[i]; a[i + 8] = hv[i]; }
    return a;
}

__device__ __forceinline__ v16h load_frag_b(const half_t* __restrict__ Wt, int Kpad,
                                            int n0, int k0) {
    int lane = threadIdx.x & 31;
    int hi   = lane >> 4;
    int n    = n0 + (lane & 15);
    const half_t* p = Wt + (size_t)n * Kpad + k0 + hi * 8;
    v8h lo = *(const v8h*)p;
    v8h hv = *(const v8h*)(p + 16);
    v16h b;
#pragma unroll
    for (int i = 0; i < 8; ++i) { b[i] = lo[i]; b[i + 8] = hv[i]; }
    return b;
}

// Template-recursive N-tile loop so the reuse-A hint is a frontend constant:
// all NT WMMAs share the A fragment and are emitted as identical back-to-back
// instructions, so tiles T>0 may legally set the reuse-A OPSEL hint.
template <int T, int NT>
__device__ __forceinline__ void wmma_tiles(const v16h& a,
                                           const half_t* __restrict__ Wt, int Kpad,
                                           int k0, v8f (&acc)[NT]) {
    if constexpr (T < NT) {
        v16h b = load_frag_b(Wt, Kpad, T * 16, k0);
        acc[T] = __builtin_amdgcn_wmma_f32_16x16x32_f16(
            false, a, false, b, (short)0, acc[T],
            /*reuse_a=*/(T > 0), /*reuse_b=*/false);
        wmma_tiles<T + 1, NT>(a, Wt, Kpad, k0, acc);
    }
}

// ---------------------------------------------------------------------------
// WMMA GEMM: out[ROWS x (NT*16)] = relu(A[ROWS x K] * W[K x N] + bias)
// 8 waves/block, 16 rows/wave, 128 rows/block -> grid.x = ROWS/128 = 256.
// SPLIT: k < ksplit reads A1[row>>rshift][k] (per-batch context broadcast),
//        k >= ksplit reads A2[row][k-ksplit]  (fused concat, layer 3a only).
// ---------------------------------------------------------------------------
template <int NT, bool SPLIT>
__global__ __launch_bounds__(256)
void gemm_wmma(const half_t* __restrict__ A2, int lda2,
               const half_t* __restrict__ A1, int lda1, int rshift, int ksplit,
               const half_t* __restrict__ Wt, int Kpad,
               const float*  __restrict__ bias, int nbias,
               half_t* __restrict__ outH, float* __restrict__ outF, int ldo,
               int K, int relu) {
    int wave = threadIdx.x >> 5;
    int lane = threadIdx.x & 31;
    int row0 = blockIdx.x * 128 + wave * 16;

    v8f zero = {};
    v8f acc[NT];
#pragma unroll
    for (int t = 0; t < NT; ++t) acc[t] = zero;

    for (int k0 = 0; k0 < K; k0 += 32) {
        v16h a = load_frag_a<SPLIT>(A2, lda2, A1, lda1, rshift, ksplit, row0, k0);

        if (k0 + 32 < K) {  // prefetch next A tile -> global_prefetch_b8
            int kn  = k0 + 32 + ((lane >> 4) << 3);
            int row = row0 + (lane & 15);
            const half_t* pf = (SPLIT && kn < ksplit)
                ? A1 + (size_t)(row >> rshift) * lda1 + kn
                : A2 + (size_t)row * lda2 + (SPLIT ? kn - ksplit : kn);
            __builtin_prefetch(pf, 0, 0);
        }

        wmma_tiles<0, NT>(a, Wt, Kpad, k0, acc);
    }

    // 32-bit C/D 16x16 layout: VGPR r -> (M = r + 8*hi, N = lane&15).
    int hi = lane >> 4, nl = lane & 15;
#pragma unroll
    for (int t = 0; t < NT; ++t) {
        int n = t * 16 + nl;
        float bv = (n < nbias) ? bias[n] : 0.0f;
#pragma unroll
        for (int r = 0; r < 8; ++r) {
            int m = r + hi * 8;
            float v = acc[t][r] + bv;
            if (relu) v = fmaxf(v, 0.0f);
            size_t o = (size_t)(row0 + m) * ldo + n;
            if (outH) outH[o] = (half_t)v;
            else      outF[o] = v;
        }
    }
}

// ---------------------------------------------------------------------------
// Weight pack: Wt[n][k] = (half)W[k][n], zero padded to (Npad, Kpad).
// ---------------------------------------------------------------------------
__global__ void pack_w(const float* __restrict__ W, int Kin, int Nin,
                       half_t* __restrict__ Wt, int Kpad, int Npad) {
    int i = blockIdx.x * blockDim.x + threadIdx.x;
    if (i >= Kpad * Npad) return;
    int k = i % Kpad, n = i / Kpad;
    float v = (k < Kin && n < Nin) ? W[(size_t)k * Nin + n] : 0.0f;
    Wt[(size_t)n * Kpad + k] = (half_t)v;
}

// A0[row][0..31] = [pos(3) | x(6) | zeros]  (features = concat(pos, x))
__global__ void pack_a0(const float* __restrict__ x, const float* __restrict__ pos,
                        half_t* __restrict__ A0) {
    int row = blockIdx.x * blockDim.x + threadIdx.x;
    if (row >= ROWS) return;
    half_t* o = A0 + (size_t)row * 32;
#pragma unroll
    for (int i = 0; i < 3; ++i) o[i] = (half_t)pos[row * 3 + i];
#pragma unroll
    for (int i = 0; i < 6; ++i) o[3 + i] = (half_t)x[row * 6 + i];
#pragma unroll
    for (int i = 9; i < 32; ++i) o[i] = (half_t)0.0f;
}

// Per-batch channel max over f2 (ROWS x 256) -> ctxH (8 x 256) f16.
__global__ __launch_bounds__(256)
void maxpool_kernel(const half_t* __restrict__ f2, half_t* __restrict__ ctxH) {
    int b = blockIdx.x;    // 8
    int c = threadIdx.x;   // 256
    const half_t* p = f2 + (size_t)b * NPTS * 256 + c;
    float m = -1e30f;
    for (int i = 0; i < NPTS; ++i) m = fmaxf(m, (float)p[(size_t)i * 256]);
    ctxH[b * 256 + c] = (half_t)m;
}

// ---------------------------------------------------------------------------
// kNN (k=3) interpolate: per target point, scan this batch's 4096 source
// points through an LDS tile, keep top-3 nearest, weighted-average logits.
// ---------------------------------------------------------------------------
#define PTILE 512
__global__ __launch_bounds__(256)
void knn_kernel(const float* __restrict__ pos,     // source (B*N, 3)
                const float* __restrict__ posy,    // target (B*M, 3)
                const float* __restrict__ logits,  // (B*N, 16) padded f32
                float* __restrict__ out) {         // (B*M, 7)
    __shared__ float4 spx[PTILE];
    int b  = blockIdx.x >> 4;           // 16 chunks of 256 targets per batch
    int mc = blockIdx.x & 15;
    int m  = mc * 256 + threadIdx.x;
    int gm = b * NPTS + m;

    float py0 = posy[gm * 3 + 0], py1 = posy[gm * 3 + 1], py2 = posy[gm * 3 + 2];
    float d0 = 1e30f, d1 = 1e30f, d2 = 1e30f;
    int   i0 = 0,     i1 = 0,     i2 = 0;

    for (int t = 0; t < NPTS; t += PTILE) {
        __syncthreads();
        for (int j = threadIdx.x; j < PTILE; j += 256) {
            int gi = b * NPTS + t + j;
            spx[j] = make_float4(pos[gi * 3 + 0], pos[gi * 3 + 1], pos[gi * 3 + 2], 0.0f);
        }
        __syncthreads();
        for (int j = 0; j < PTILE; ++j) {
            float4 p = spx[j];
            float dx = py0 - p.x, dy = py1 - p.y, dz = py2 - p.z;
            float d = dx * dx + dy * dy + dz * dz;
            int idx = t + j;
            if (d < d2) {
                if (d < d1) {
                    d2 = d1; i2 = i1;
                    if (d < d0) { d1 = d0; i1 = i0; d0 = d; i0 = idx; }
                    else        { d1 = d;  i1 = idx; }
                } else { d2 = d; i2 = idx; }
            }
        }
    }

    float w0 = 1.0f / fmaxf(d0, 1e-16f);
    float w1 = 1.0f / fmaxf(d1, 1e-16f);
    float w2 = 1.0f / fmaxf(d2, 1e-16f);
    float ws = w0 + w1 + w2;
    const float* l0 = logits + (size_t)(b * NPTS + i0) * 16;
    const float* l1 = logits + (size_t)(b * NPTS + i1) * 16;
    const float* l2 = logits + (size_t)(b * NPTS + i2) * 16;
    float* o = out + (size_t)gm * 7;
#pragma unroll
    for (int c = 0; c < 7; ++c)
        o[c] = (w0 * l0[c] + w1 * l1[c] + w2 * l2[c]) / ws;
}

// ---------------------------------------------------------------------------
// Host launch: pack weights/inputs, run GEMM chain, max-pool, kNN.
// ---------------------------------------------------------------------------
extern "C" void kernel_launch(void* const* d_in, const int* in_sizes, int n_in,
                              void* d_out, int out_size, void* d_ws, size_t ws_size,
                              hipStream_t stream) {
    (void)in_sizes; (void)n_in; (void)out_size; (void)ws_size;

    const float* x    = (const float*)d_in[0];
    const float* pos  = (const float*)d_in[1];
    const float* posy = (const float*)d_in[2];
    const float* m1W1 = (const float*)d_in[5];  const float* m1b1 = (const float*)d_in[6];
    const float* m1W2 = (const float*)d_in[7];  const float* m1b2 = (const float*)d_in[8];
    const float* m2W1 = (const float*)d_in[9];  const float* m2b1 = (const float*)d_in[10];
    const float* m2W2 = (const float*)d_in[11]; const float* m2b2 = (const float*)d_in[12];
    const float* m2W3 = (const float*)d_in[13]; const float* m2b3 = (const float*)d_in[14];
    const float* m3W1 = (const float*)d_in[15]; const float* m3b1 = (const float*)d_in[16];
    const float* m3W2 = (const float*)d_in[17]; const float* m3b2 = (const float*)d_in[18];
    const float* linW = (const float*)d_in[19]; const float* linb = (const float*)d_in[20];
    float* out = (float*)d_out;

    char* ws = (char*)d_ws;
    const size_t MB = 1024ull * 1024ull;
    // Weight / context region [0, 1MB)
    half_t* W1t   = (half_t*)(ws + 0);        // 64x32    (4 KB)
    half_t* W2t   = (half_t*)(ws + 4096);     // 64x64    (8 KB)
    half_t* m2W1t = (half_t*)(ws + 12288);    // 64x64    (8 KB)
    half_t* m2W2t = (half_t*)(ws + 20480);    // 128x64   (16 KB)
    half_t* m2W3t = (half_t*)(ws + 36864);    // 256x128  (64 KB)
    half_t* m3W1t = (half_t*)(ws + 102400);   // 256x320  (160 KB)
    half_t* m3W2t = (half_t*)(ws + 266240);   // 128x256  (64 KB)
    half_t* linWt = (half_t*)(ws + 331776);   // 16x128   (4 KB)
    half_t* ctxH  = (half_t*)(ws + 335872);   // 8x256    (4 KB)
    // Activations (f16), sequential reuse of regions:
    half_t* A0   = (half_t*)(ws + 1 * MB);    // 32768x32   2 MB
    half_t* f1   = (half_t*)(ws + 3 * MB);    // 32768x64   4 MB (live until L3a)
    half_t* h1   = (half_t*)(ws + 7 * MB);    // 32768x64   4 MB  } region reused:
    half_t* t2b  = (half_t*)(ws + 7 * MB);    // 32768x128  8 MB  } h1 -> t2b -> f3
    half_t* f3   = (half_t*)(ws + 7 * MB);    // 32768x128  8 MB  }
    half_t* t2a  = (half_t*)(ws + 15 * MB);   // 32768x64   4 MB  } region reused:
    half_t* t3   = (half_t*)(ws + 15 * MB);   // 32768x256 16 MB  } t2a -> t3
    half_t* f2   = (half_t*)(ws + 31 * MB);   // 32768x256 16 MB
    float*  logp = (float*) (ws + 47 * MB);   // 32768x16   2 MB (padded logits)

    // ---- pack weights (transpose + pad) and input features --------------
    auto packw = [&](const float* W, int Kin, int Nin, half_t* Wt, int Kpad, int Npad) {
        int tot = Kpad * Npad;
        pack_w<<<(tot + 255) / 256, 256, 0, stream>>>(W, Kin, Nin, Wt, Kpad, Npad);
    };
    packw(m1W1,   9,  64, W1t,    32,  64);
    packw(m1W2,  64,  64, W2t,    64,  64);
    packw(m2W1,  64,  64, m2W1t,  64,  64);
    packw(m2W2,  64, 128, m2W2t,  64, 128);
    packw(m2W3, 128, 256, m2W3t, 128, 256);
    packw(m3W1, 320, 256, m3W1t, 320, 256);
    packw(m3W2, 256, 128, m3W2t, 256, 128);
    packw(linW, 128,   7, linWt, 128,  16);
    pack_a0<<<ROWS / 256, 256, 0, stream>>>(x, pos, A0);

    const int GB = ROWS / 128;  // 256 blocks, 8 waves x 16 rows each

    // ---- MLP1: 9->64->64 -------------------------------------------------
    gemm_wmma<4, false><<<GB, 256, 0, stream>>>(A0, 32, nullptr, 0, 0, 0,
                                                W1t, 32, m1b1, 64, h1, nullptr, 64, 32, 1);
    gemm_wmma<4, false><<<GB, 256, 0, stream>>>(h1, 64, nullptr, 0, 0, 0,
                                                W2t, 64, m1b2, 64, f1, nullptr, 64, 64, 1);
    // ---- MLP2: 64->64->128->256 -----------------------------------------
    gemm_wmma<4, false><<<GB, 256, 0, stream>>>(f1, 64, nullptr, 0, 0, 0,
                                                m2W1t, 64, m2b1, 64, t2a, nullptr, 64, 64, 1);
    gemm_wmma<8, false><<<GB, 256, 0, stream>>>(t2a, 64, nullptr, 0, 0, 0,
                                                m2W2t, 64, m2b2, 128, t2b, nullptr, 128, 64, 1);
    gemm_wmma<16, false><<<GB, 256, 0, stream>>>(t2b, 128, nullptr, 0, 0, 0,
                                                 m2W3t, 128, m2b3, 256, f2, nullptr, 256, 128, 1);
    // ---- global max pool -> context (8x256) ------------------------------
    maxpool_kernel<<<NB, 256, 0, stream>>>(f2, ctxH);
    // ---- MLP3: concat(context,f1)=320 -> 256 -> 128  (concat fused in A) -
    gemm_wmma<16, true><<<GB, 256, 0, stream>>>(f1, 64, ctxH, 256, 12, 256,
                                                m3W1t, 320, m3b1, 256, t3, nullptr, 256, 320, 1);
    gemm_wmma<8, false><<<GB, 256, 0, stream>>>(t3, 256, nullptr, 0, 0, 0,
                                                m3W2t, 256, m3b2, 128, f3, nullptr, 128, 256, 1);
    // ---- classifier: 128 -> 7 (padded to 16, f32 out, no relu) -----------
    gemm_wmma<1, false><<<GB, 256, 0, stream>>>(f3, 128, nullptr, 0, 0, 0,
                                                linWt, 128, linb, 7, nullptr, logp, 16, 128, 0);
    // ---- kNN (k=3) interpolation -----------------------------------------
    knn_kernel<<<NB * 16, 256, 0, stream>>>(pos, posy, logp, out);
}
